// MambaLM_56650618634256
// MI455X (gfx1250) — matrix operand order repaired and checked by reading, not compiled, not run on hardware
//
#include <hip/hip_runtime.h>
#include <hip/hip_bf16.h>

// ---------- problem constants ----------
#define V_  32000
#define D_  768
#define H_  12
#define N_  64
#define K_  4
#define LYR_ 4
#define B_  2
#define L_  2048
#define M_  (B_ * L_)          // 4096 tokens
#define PP_ (2 * H_ * N_)      // 1536
#define KD_ 768                // shared K dimension of every GEMM

typedef __attribute__((ext_vector_type(16))) __bf16 v16bf;
typedef __attribute__((ext_vector_type(8)))  float  v8f;

union FragBF {
    v16bf v;
    uint4 q[2];
};

__device__ __forceinline__ unsigned short f2bf(float f) {
    unsigned int u = __float_as_uint(f);
    unsigned int r = u + 0x7FFFu + ((u >> 16) & 1u);   // round-to-nearest-even
    return (unsigned short)(r >> 16);
}

__device__ __forceinline__ float sigmoidf_(float x) {
    return 1.0f / (1.0f + __expf(-x));
}

// ---------- elementwise / gather kernels ----------
__global__ void k_embed(const int* __restrict__ tok, const float* __restrict__ emb,
                        float* __restrict__ x) {
    int i = blockIdx.x * blockDim.x + threadIdx.x;
    if (i >= M_ * D_) return;
    int t = i / D_;
    int d = i - t * D_;
    x[i] = emb[(size_t)tok[t] * D_ + d];
}

__global__ void k_f32_to_bf16(const float* __restrict__ in, unsigned short* __restrict__ out,
                              int n) {
    int i = blockIdx.x * blockDim.x + threadIdx.x;
    if (i < n) out[i] = f2bf(in[i]);
}

// transpose fp32 (Kd x N) -> bf16 (N x Kd); coalesced reads
__global__ void k_transpose_bf16(const float* __restrict__ w, unsigned short* __restrict__ wt,
                                 int Kd, int N) {
    int i = blockIdx.x * blockDim.x + threadIdx.x;
    if (i >= Kd * N) return;
    int k = i / N;
    int n = i - k * N;
    wt[(size_t)n * Kd + k] = f2bf(w[i]);
}

// ---------- rmsnorm: one block per token, writes f32 + bf16 ----------
__global__ __launch_bounds__(256)
void k_rmsnorm(const float* __restrict__ x, const float* __restrict__ w,
               float* __restrict__ out_f32, unsigned short* __restrict__ out_bf) {
    int t   = blockIdx.x;
    int tid = threadIdx.x;
    __shared__ float red[256];
    float v[3];
    float ss = 0.f;
#pragma unroll
    for (int i = 0; i < 3; ++i) {
        int d = tid + i * 256;
        float a = x[(size_t)t * D_ + d];
        v[i] = a;
        ss += a * a;
    }
    red[tid] = ss;
    __syncthreads();
    for (int o = 128; o > 0; o >>= 1) {
        if (tid < o) red[tid] += red[tid + o];
        __syncthreads();
    }
    float inv = 1.0f / sqrtf(red[0] * (1.0f / D_) + 1e-6f);
#pragma unroll
    for (int i = 0; i < 3; ++i) {
        int d = tid + i * 256;
        float o = w[d] * v[i] * inv;
        out_f32[(size_t)t * D_ + d] = o;
        out_bf[(size_t)t * D_ + d]  = f2bf(o);
    }
}

// ---------- mix + rmsnorm (extra_norm), bf16 out for the out-projection ----------
__global__ __launch_bounds__(256)
void k_mix_rmsnorm(const float* __restrict__ gate, const float* __restrict__ ssm,
                   const float* __restrict__ xnorm, const float* __restrict__ w,
                   unsigned short* __restrict__ out_bf) {
    int t   = blockIdx.x;
    int tid = threadIdx.x;
    __shared__ float red[256];
    float m[3];
    float ss = 0.f;
#pragma unroll
    for (int i = 0; i < 3; ++i) {
        int d = tid + i * 256;
        size_t idx = (size_t)t * D_ + d;
        float g = gate[idx];
        float v = g * ssm[idx] + (1.0f - g) * xnorm[idx];
        m[i] = v;
        ss += v * v;
    }
    red[tid] = ss;
    __syncthreads();
    for (int o = 128; o > 0; o >>= 1) {
        if (tid < o) red[tid] += red[tid + o];
        __syncthreads();
    }
    float inv = 1.0f / sqrtf(red[0] * (1.0f / D_) + 1e-6f);
#pragma unroll
    for (int i = 0; i < 3; ++i) {
        int d = tid + i * 256;
        out_bf[(size_t)t * D_ + d] = f2bf(w[d] * m[i] * inv);
    }
}

// ---------- dt projection + delta = exp(-dt) ----------
__global__ void k_dt_delta(const float* __restrict__ xn, const float* __restrict__ w,
                           const float* __restrict__ bvec, float* __restrict__ delta) {
    int i = blockIdx.x * blockDim.x + threadIdx.x;   // over M_*H_
    if (i >= M_ * H_) return;
    int h = i % H_;
    int t = i / H_;
    const float* xr = xn + (size_t)t * D_;
    float acc = bvec[h];
    for (int d = 0; d < D_; ++d) acc += xr[d] * w[d * H_ + h];
    float dt = 1e-4f + (0.1f - 1e-4f) * sigmoidf_(acc);
    delta[i] = __expf(-dt);
}

// ---------- depthwise causal conv (K=4) + SiLU, bf16 out ----------
__global__ void k_conv_silu(const float* __restrict__ xn, const float* __restrict__ cw,
                            const float* __restrict__ cb, unsigned short* __restrict__ out_bf) {
    int i = blockIdx.x * blockDim.x + threadIdx.x;   // over M_*D_
    if (i >= M_ * D_) return;
    int d = i % D_;
    int t = i / D_;
    int l = t % L_;
    float acc = cb[d];
#pragma unroll
    for (int k = 0; k < K_; ++k) {
        int ls = l - (K_ - 1) + k;
        if (ls >= 0) acc += cw[d * K_ + k] * xn[(size_t)(t - (K_ - 1) + k) * D_ + d];
    }
    out_bf[i] = f2bf(acc * sigmoidf_(acc));
}

// ---------- selective scan: one thread per (b,h,n) chain ----------
__global__ void k_scan(const float* __restrict__ delta, const float* __restrict__ params,
                       float* __restrict__ ssm) {
    int t = blockIdx.x * blockDim.x + threadIdx.x;
    if (t >= B_ * H_ * N_) return;
    int n = t % N_;
    int h = (t / N_) % H_;
    int b = t / (N_ * H_);
    float hs = 0.f;
    for (int l = 0; l < L_; ++l) {
        size_t tok = (size_t)b * L_ + l;
        float dlt = delta[tok * H_ + h];
        size_t pb = (tok * H_ + h) * (size_t)(2 * N_);
        float Bt = params[pb + n];
        float Ct = params[pb + N_ + n];
        hs = dlt * hs + Bt;
        ssm[tok * D_ + h * N_ + n] = Ct * hs;
    }
}

// ---------- WMMA bf16 GEMM ----------
// C[M x N] = A[M x 768] * BT[N x 768]^T (+bias)(+resid)(act). ACT: 0 none, 1 sigmoid.
// Block = 8 waves sharing one 32-row M-tile staged in LDS; each wave owns a 32x64 tile
// (8 accumulators). B fragments double-buffered in two STATICALLY-NAMED register sets
// (bA/bB) so nothing is dynamically indexed (no scratch spill); K-loop advances 64/iter
// and is fully unrolled so all offsets are immediates.
#define LDS_STRIDE (KD_ + 8)   // 776 bf16 per row = 97 uint4 (skew kills bank conflicts)

template <int ACT>
__global__ __launch_bounds__(256)
void k_gemm_bf16(const unsigned short* __restrict__ A,
                 const unsigned short* __restrict__ BT,
                 const float* __restrict__ bias,
                 const float* __restrict__ resid,
                 float* __restrict__ C,
                 int M, int N) {
    __shared__ unsigned short As[32 * LDS_STRIDE];   // ~48.5 KB of the 320 KB WGP LDS

    int tid = threadIdx.x;
    int m0g = blockIdx.y * 32;

    // cooperative stage of A tile: 32 rows x 768 bf16 (96 uint4 per row, stride 97)
    {
        const uint4* src = (const uint4*)(A + (size_t)m0g * KD_);
        uint4* dst = (uint4*)As;
#pragma unroll
        for (int it = 0; it < 12; ++it) {
            int c = tid + it * 256;            // c in [0, 32*96)
            int r = c / 96;
            int o = c - r * 96;
            dst[r * 97 + o] = src[r * 96 + o];
        }
    }
    __syncthreads();

    int wave = tid >> 5;
    int n0 = blockIdx.x * 512 + wave * 64;
    if (n0 >= N) return;                       // wave-uniform: EXEC stays all-ones

    int lane = tid & 31;
    int sel  = lane >> 4;                      // 0: k 0-7,16-23 of each 32-k step; 1: k 8-15,24-31
    int l16  = lane & 15;

    const unsigned short* a0p = As + l16 * LDS_STRIDE + sel * 8;         // rows 0..15
    const unsigned short* a1p = As + (16 + l16) * LDS_STRIDE + sel * 8;  // rows 16..31
    const unsigned short* bbase = BT + (size_t)(n0 + l16) * KD_ + sel * 8;

    v8f acc[2][4] = {};
    FragBF bA[4], bB[4];

    auto loadB = [&](FragBF (&dst)[4], int kb) {
#pragma unroll
        for (int j = 0; j < 4; ++j) {
            const unsigned short* bp = bbase + (size_t)(j * 16) * KD_ + kb;
            dst[j].q[0] = *(const uint4*)(bp);
            dst[j].q[1] = *(const uint4*)(bp + 16);
        }
    };
    auto mulAcc = [&](FragBF (&bf)[4], int kb) {
        FragBF af0, af1;
        af0.q[0] = *(const uint4*)(a0p + kb);
        af0.q[1] = *(const uint4*)(a0p + kb + 16);
        af1.q[0] = *(const uint4*)(a1p + kb);
        af1.q[1] = *(const uint4*)(a1p + kb + 16);
#pragma unroll
        for (int j = 0; j < 4; ++j)
            acc[0][j] = __builtin_amdgcn_wmma_f32_16x16x32_bf16(
                false, af0.v, false, bf[j].v, (short)0, acc[0][j], false, false);
#pragma unroll
        for (int j = 0; j < 4; ++j)
            acc[1][j] = __builtin_amdgcn_wmma_f32_16x16x32_bf16(
                false, af1.v, false, bf[j].v, (short)0, acc[1][j], false, false);
    };

    loadB(bA, 0);
#pragma unroll
    for (int kb = 0; kb < KD_; kb += 64) {     // 12 fully-unrolled iterations
        loadB(bB, kb + 32);                    // issue next K-step's loads first
        mulAcc(bA, kb);                        // math depends only on older loads
        if (kb + 64 < KD_) loadB(bA, kb + 64);
        mulAcc(bB, kb + 32);
    }

    // epilogue: C/D layout — VGPR i holds M = i (lanes 0-15) / M = i+8 (lanes 16-31)
#pragma unroll
    for (int g = 0; g < 2; ++g) {
        int mrow = m0g + g * 16;
#pragma unroll
        for (int j = 0; j < 4; ++j) {
            int col  = n0 + j * 16 + l16;
            float bv = bias ? bias[col] : 0.0f;
#pragma unroll
            for (int i = 0; i < 8; ++i) {
                int row = mrow + i + sel * 8;
                size_t idx = (size_t)row * N + col;
                float v = acc[g][j][i] + bv;
                if (ACT == 1) v = sigmoidf_(v);
                if (resid) v += resid[idx];
                C[idx] = v;
            }
        }
    }
}

// ---------- host orchestration ----------
static inline int cdiv(long long a, long long b) { return (int)((a + b - 1) / b); }

extern "C" void kernel_launch(void* const* d_in, const int* in_sizes, int n_in,
                              void* d_out, int out_size, void* d_ws, size_t ws_size,
                              hipStream_t stream) {
    const int*   tokens       = (const int*)d_in[0];
    const float* emb          = (const float*)d_in[1];
    const float* norm_w       = (const float*)d_in[2];
    const float* dt_w         = (const float*)d_in[3];
    const float* dt_b         = (const float*)d_in[4];
    const float* pp_w         = (const float*)d_in[5];
    const float* pp_b         = (const float*)d_in[6];
    const float* conv_w       = (const float*)d_in[7];
    const float* conv_b       = (const float*)d_in[8];
    const float* gate_w       = (const float*)d_in[9];
    const float* gate_b       = (const float*)d_in[10];
    const float* out_w        = (const float*)d_in[11];
    const float* out_b        = (const float*)d_in[12];
    const float* extra_norm_w = (const float*)d_in[13];
    const float* final_norm_w = (const float*)d_in[14];
    float* logits = (float*)d_out;

    // workspace bump allocator
    char* p = (char*)d_ws;
    auto alloc = [&](size_t bytes) -> void* {
        void* r = (void*)p;
        p += (bytes + 255) & ~(size_t)255;
        return r;
    };
    float*          xA       = (float*)alloc((size_t)M_ * D_ * 4);
    float*          xB       = (float*)alloc((size_t)M_ * D_ * 4);
    float*          xnorm    = (float*)alloc((size_t)M_ * D_ * 4);
    unsigned short* xnorm_bf = (unsigned short*)alloc((size_t)M_ * D_ * 2);
    unsigned short* xconv_bf = (unsigned short*)alloc((size_t)M_ * D_ * 2);
    unsigned short* mixed_bf = (unsigned short*)alloc((size_t)M_ * D_ * 2);
    float*          params   = (float*)alloc((size_t)M_ * PP_ * 4);
    float*          delta    = (float*)alloc((size_t)M_ * H_ * 4);
    float*          gatebuf  = (float*)alloc((size_t)M_ * D_ * 4);
    float*          ssm      = (float*)alloc((size_t)M_ * D_ * 4);
    unsigned short* emb_bf   = (unsigned short*)alloc((size_t)V_ * D_ * 2);
    unsigned short* wT       = (unsigned short*)alloc((size_t)PP_ * D_ * 2); // max transpose

    const int TB = 256;

    // embedding gather + bf16 copy of emb (B operand of the logits GEMM, K-contiguous as-is)
    k_embed<<<cdiv((long long)M_ * D_, TB), TB, 0, stream>>>(tokens, emb, xA);
    k_f32_to_bf16<<<cdiv((long long)V_ * D_, TB), TB, 0, stream>>>(emb, emb_bf, V_ * D_);

    auto gemm_grid = [](int N) { return dim3((unsigned)cdiv(N, 512), (unsigned)(M_ / 32)); };

    float* cur = xA;
    float* nxt = xB;
    for (int l = 0; l < LYR_; ++l) {
        // rmsnorm -> xnorm (f32 + bf16)
        k_rmsnorm<<<M_, 256, 0, stream>>>(cur, norm_w + (size_t)l * D_, xnorm, xnorm_bf);

        // pp projection: params = xnorm @ pp_w + pp_b   (4096 x 1536)
        k_transpose_bf16<<<cdiv((long long)D_ * PP_, TB), TB, 0, stream>>>(
            pp_w + (size_t)l * D_ * PP_, wT, D_, PP_);
        k_gemm_bf16<0><<<gemm_grid(PP_), 256, 0, stream>>>(
            xnorm_bf, wT, pp_b + (size_t)l * PP_, nullptr, params, M_, PP_);

        // dt projection + delta
        k_dt_delta<<<cdiv((long long)M_ * H_, 128), 128, 0, stream>>>(
            xnorm, dt_w + (size_t)l * D_ * H_, dt_b + (size_t)l * H_, delta);

        // depthwise conv + SiLU -> bf16
        k_conv_silu<<<cdiv((long long)M_ * D_, TB), TB, 0, stream>>>(
            xnorm, conv_w + (size_t)l * D_ * K_, conv_b + (size_t)l * D_, xconv_bf);

        // gate = sigmoid(x_conv @ gate_w + gate_b)   (4096 x 768)
        k_transpose_bf16<<<cdiv((long long)D_ * D_, TB), TB, 0, stream>>>(
            gate_w + (size_t)l * D_ * D_, wT, D_, D_);
        k_gemm_bf16<1><<<gemm_grid(D_), 256, 0, stream>>>(
            xconv_bf, wT, gate_b + (size_t)l * D_, nullptr, gatebuf, M_, D_);

        // selective scan over L (1536 independent chains)
        k_scan<<<cdiv(B_ * H_ * N_, 256), 256, 0, stream>>>(delta, params, ssm);

        // mixed = gate*ssm + (1-gate)*xnorm, then rmsnorm(extra) -> bf16
        k_mix_rmsnorm<<<M_, 256, 0, stream>>>(
            gatebuf, ssm, xnorm, extra_norm_w + (size_t)l * D_, mixed_bf);

        // y = mixed_norm @ out_w + out_b + x(residual)
        k_transpose_bf16<<<cdiv((long long)D_ * D_, TB), TB, 0, stream>>>(
            out_w + (size_t)l * D_ * D_, wT, D_, D_);
        k_gemm_bf16<0><<<gemm_grid(D_), 256, 0, stream>>>(
            mixed_bf, wT, out_b + (size_t)l * D_, cur, nxt, M_, D_);

        float* tmp = cur; cur = nxt; nxt = tmp;
    }

    // final rmsnorm, then logits = x @ emb.T   (4096 x 32000, K = 768)
    k_rmsnorm<<<M_, 256, 0, stream>>>(cur, final_norm_w, xnorm, xnorm_bf);
    k_gemm_bf16<0><<<gemm_grid(V_), 256, 0, stream>>>(
        xnorm_bf, emb_bf, nullptr, nullptr, logits, M_, V_);

    (void)in_sizes; (void)n_in; (void)out_size; (void)ws_size;
}
